// FullAttention_50646254355303
// MI455X (gfx1250) — compile-verified, matrix operand order checked by reading, not run on hardware
//
#include <hip/hip_runtime.h>
#include <math.h>

// ---------------------------------------------------------------- types
typedef __attribute__((ext_vector_type(16))) __bf16 v16bf;
typedef __attribute__((ext_vector_type(8)))  float  v8f;
typedef int v4i_vec __attribute__((vector_size(16)));   // matches builtin param

// ---------------------------------------------------------------- problem constants
constexpr int CB   = 2;      // batch
constexpr int CS   = 2048;   // seq
constexpr int CHID = 2048;   // hidden
constexpr int CH   = 16;     // q heads
constexpr int CKV  = 4;      // kv heads
constexpr int CD   = 128;    // head dim
constexpr int CM   = CB * CS;        // 4096 token rows
constexpr int CNQ  = 2 * CH * CD;    // 4096 (q + gate)
constexpr int CNK  = CKV * CD;       // 512

// ---------------------------------------------------------------- async global->LDS copy
#if defined(__has_builtin)
#  if __has_builtin(__builtin_amdgcn_global_load_async_to_lds_b128)
#    define ASYNC_VIA_BUILTIN 1
#  endif
#  if __has_builtin(__builtin_amdgcn_s_wait_asynccnt)
#    define WAITASYNC_VIA_BUILTIN 1
#  endif
#endif

// copy 16 bytes (8 bf16) global -> LDS, tracked by ASYNCcnt
__device__ inline void async_cp16(__bf16* dst_lds, const __bf16* src_glb) {
#ifdef ASYNC_VIA_BUILTIN
    __builtin_amdgcn_global_load_async_to_lds_b128(
        (__attribute__((address_space(1))) v4i_vec*)src_glb,
        (__attribute__((address_space(3))) v4i_vec*)dst_lds,
        0, 0);
#else
    unsigned lds_off =
        (unsigned)(unsigned long long)(__attribute__((address_space(3))) __bf16*)dst_lds;
    unsigned long long ga = (unsigned long long)src_glb;
    asm volatile("global_load_async_to_lds_b128 %0, %1, off"
                 :: "v"(lds_off), "v"(ga) : "memory");
#endif
}

__device__ inline void wait_async0() {
#ifdef WAITASYNC_VIA_BUILTIN
    __builtin_amdgcn_s_wait_asynccnt(0);
#else
    asm volatile("s_wait_asynccnt 0x0" ::: "memory");
#endif
}

// ---------------------------------------------------------------- WMMA fragment load
// A-frag (16x32 bf16, lane=row) / B-frag (32x16, lane=col): lane l -> r=l&15,
// hi=l>>4; frag elems 0..7 = K[8*hi..+7], elems 8..15 = K[16+8*hi..+7].
__device__ inline v16bf load_frag(const __bf16* base, int stride_elems) {
    int lane = threadIdx.x & 31;
    int r = lane & 15, hi = lane >> 4;
    const __bf16* p = base + (size_t)r * stride_elems + 8 * hi;
    v16bf f;
    ((uint4*)&f)[0] = *(const uint4*)(p);
    ((uint4*)&f)[1] = *(const uint4*)(p + 16);
    return f;
}

// ---------------------------------------------------------------- casts
__global__ __launch_bounds__(256) void cast_bf16_kernel(const float* __restrict__ s,
                                                        __bf16* __restrict__ d, int n) {
    int i = blockIdx.x * 256 + threadIdx.x;
    if (i < n) d[i] = (__bf16)s[i];
}

// src: K x N row-major fp32  ->  dst: N x K row-major bf16 (B^T for GEMM)
__global__ __launch_bounds__(256) void cast_transpose_kernel(const float* __restrict__ s,
                                                             __bf16* __restrict__ d,
                                                             int K, int N) {
    int i = blockIdx.x * 256 + threadIdx.x;
    if (i < K * N) {
        int k = i / N, n = i - k * N;
        d[(size_t)n * K + k] = (__bf16)s[i];
    }
}

// ---------------------------------------------------------------- GEMM  C[M,N] = A[M,K] * Bt[N,K]^T
// 128x128 tile, K-step 32, double-buffered async LDS staging.
// 8 waves: wave grid 4 (rows of 32) x 2 (cols of 64).
__global__ __launch_bounds__(256) void gemm_bf16_wmma(const __bf16* __restrict__ A,
                                                      const __bf16* __restrict__ Bt,
                                                      float* __restrict__ C,
                                                      int M, int N, int K) {
    __shared__ alignas(16) __bf16 lA[2][128 * 32];
    __shared__ alignas(16) __bf16 lB[2][128 * 32];

    const int tid  = threadIdx.x;
    const int wave = tid >> 5;
    const int lane = tid & 31;
    const int tile_m = blockIdx.y * 128;
    const int tile_n = blockIdx.x * 128;
    const int wm = (wave & 3) * 32;   // wave row offset in tile
    const int wn = (wave >> 2) * 64;  // wave col offset in tile

    v8f acc[2][4] = {};

    const int ar = tid >> 1;          // 0..127 row of tile
    const int as = (tid & 1) * 16;    // 16-elem segment
    const __bf16* gA = A  + (size_t)(tile_m + ar) * K + as;
    const __bf16* gB = Bt + (size_t)(tile_n + ar) * K + as;

    auto stage = [&](int buf, int k0) {
        __bf16* dA = lA[buf] + ar * 32 + as;
        __bf16* dB = lB[buf] + ar * 32 + as;
        async_cp16(dA,     gA + k0);
        async_cp16(dA + 8, gA + k0 + 8);
        async_cp16(dB,     gB + k0);
        async_cp16(dB + 8, gB + k0 + 8);
    };

    const int nk = K >> 5;
    stage(0, 0);
    wait_async0();
    __syncthreads();

    for (int kt = 0; kt < nk; ++kt) {
        const int cur = kt & 1;
        if (kt + 1 < nk) stage(cur ^ 1, (kt + 1) << 5);

        v16bf a0 = load_frag(lA[cur] + (wm +  0) * 32, 32);
        v16bf a1 = load_frag(lA[cur] + (wm + 16) * 32, 32);
        #pragma unroll
        for (int j = 0; j < 4; ++j) {
            v16bf b = load_frag(lB[cur] + (wn + 16 * j) * 32, 32);
            acc[0][j] = __builtin_amdgcn_wmma_f32_16x16x32_bf16(
                false, a0, false, b, (short)0, acc[0][j], false, false);
            acc[1][j] = __builtin_amdgcn_wmma_f32_16x16x32_bf16(
                false, a1, false, b, (short)0, acc[1][j], false, false);
        }
        wait_async0();            // my prefetch for buf^1 has landed
        __syncthreads();          // all waves done with cur + staged next
    }

    const int hi = lane >> 4, col = lane & 15;
    #pragma unroll
    for (int i = 0; i < 2; ++i)
        #pragma unroll
        for (int j = 0; j < 4; ++j)
            #pragma unroll
            for (int v = 0; v < 8; ++v) {
                int gm = tile_m + wm + 16 * i + 8 * hi + v;
                int gn = tile_n + wn + 16 * j + col;
                C[(size_t)gm * N + gn] = acc[i][j][v];
            }
}

// ---------------------------------------------------------------- RMSNorm + RoPE + layout
__device__ inline void norm_rope_row(const float* __restrict__ src,
                                     const float* __restrict__ w,
                                     float pos, __bf16* __restrict__ dst) {
    int lane = threadIdx.x & 31;
    float v[4], nv[4];
    float ss = 0.f;
    #pragma unroll
    for (int j = 0; j < 4; ++j) { v[j] = src[lane * 4 + j]; ss += v[j] * v[j]; }
    #pragma unroll
    for (int m = 16; m; m >>= 1) ss += __shfl_xor(ss, m, 32);
    float inv = rsqrtf(ss * (1.0f / 128.0f) + 1e-6f);
    #pragma unroll
    for (int j = 0; j < 4; ++j) {
        int d = lane * 4 + j;
        nv[j] = v[j] * inv * (w[d] + 1.0f);
    }
    #pragma unroll
    for (int j = 0; j < 4; ++j) {
        int d = lane * 4 + j;
        float partner = __shfl_xor(nv[j], 16, 32);      // element at d +/- 64
        int   i   = d & 63;
        float ang = pos * __powf(1.0e6f, -(float)i * (1.0f / 64.0f));
        float c = __cosf(ang), sn = __sinf(ang);
        float out = (d < 64) ? (nv[j] * c - partner * sn)
                             : (nv[j] * c + partner * sn);
        dst[d] = (__bf16)out;
    }
}

__global__ __launch_bounds__(256) void qkv_postproc_kernel(
        const float* __restrict__ qraw,   // [M, 4096]  (H x (D q | D gate))
        const float* __restrict__ kraw,   // [M, 512]
        const float* __restrict__ vraw,   // [M, 512]
        const float* __restrict__ qn_w, const float* __restrict__ kn_w,
        __bf16* __restrict__ qb,          // [B,H,S,D]
        float*  __restrict__ gate,        // [B,H,S,D]
        __bf16* __restrict__ kb,          // [B,KV,S,D]
        __bf16* __restrict__ vb) {        // [B,KV,S,D]
    int t = blockIdx.x;                 // token = b*S + s
    int b = t / CS, s = t - b * CS;
    int wave = threadIdx.x >> 5, lane = threadIdx.x & 31;
    float pos = (float)s;

    #pragma unroll
    for (int rep = 0; rep < 2; ++rep) {
        int h = wave + 8 * rep;
        const float* src = qraw + (size_t)t * CNQ + h * (2 * CD);
        size_t ob = ((size_t)(b * CH + h) * CS + s) * CD;
        norm_rope_row(src, qn_w, pos, qb + ob);
        #pragma unroll
        for (int j = 0; j < 4; ++j) {
            int d = lane * 4 + j;
            gate[ob + d] = src[CD + d];
        }
    }
    if (wave < 4) {                       // k heads
        int kvh = wave;
        const float* src = kraw + (size_t)t * CNK + kvh * CD;
        size_t ob = ((size_t)(b * CKV + kvh) * CS + s) * CD;
        norm_rope_row(src, kn_w, pos, kb + ob);
    } else {                              // v heads (plain cast)
        int kvh = wave - 4;
        const float* src = vraw + (size_t)t * CNK + kvh * CD;
        size_t ob = ((size_t)(b * CKV + kvh) * CS + s) * CD;
        #pragma unroll
        for (int j = 0; j < 4; ++j) {
            int d = lane * 4 + j;
            vb[ob + d] = (__bf16)src[d];
        }
    }
}

// ---------------------------------------------------------------- flash attention (WMMA)
// grid (S/128, H, B); 8 waves x 16 q-rows; KV tile = 32; online softmax.
__global__ __launch_bounds__(256) void flash_attn_wmma(
        const __bf16* __restrict__ Q,    // [B,H,S,D]
        const __bf16* __restrict__ Kc,   // [B,KV,S,D]
        const __bf16* __restrict__ Vc,   // [B,KV,S,D]
        const float*  __restrict__ Gate, // [B,H,S,D]
        __bf16* __restrict__ Out) {      // [B,S,H*D]
    __shared__ alignas(16) __bf16 lK [32 * 128];   // [kv][d]
    __shared__ alignas(16) __bf16 lVt[128 * 32];   // [d][kv]
    __shared__ alignas(16) __bf16 lP [8][16 * 32]; // per-wave P staging

    const int b = blockIdx.z, h = blockIdx.y, kvh = h >> 2;
    const int tid = threadIdx.x, wave = tid >> 5, lane = tid & 31;
    const int hi = lane >> 4, col = lane & 15;
    const int qrow0 = blockIdx.x * 128 + wave * 16;
    const float scale = 0.08838834764831845f;      // 1/sqrt(128)

    const __bf16* qbase = Q + ((size_t)(b * CH + h) * CS + qrow0) * CD;
    v16bf aq[4];
    #pragma unroll
    for (int kk = 0; kk < 4; ++kk) aq[kk] = load_frag(qbase + kk * 32, CD);

    v8f  o[8] = {};
    float mrow[8], lrow[8];
    #pragma unroll
    for (int v = 0; v < 8; ++v) { mrow[v] = -1e30f; lrow[v] = 0.f; }

    const size_t kvbase = (size_t)(b * CKV + kvh) * CS;
    const int kv_end = blockIdx.x * 128 + 128;     // causal bound for block

    const int sr  = tid >> 3;            // 0..31 kv row for staging
    const int ssg = (tid & 7) * 16;      // 16-elem segment of d

    for (int kv0 = 0; kv0 < kv_end; kv0 += 32) {
        // K tile [kv][d]: async copy straight into LDS
        {
            const __bf16* ksrc = Kc + (kvbase + kv0 + sr) * CD + ssg;
            __bf16* kd = lK + sr * 128 + ssg;
            async_cp16(kd,     ksrc);
            async_cp16(kd + 8, ksrc + 8);
        }
        // V tile transposed [d][kv]: through VGPRs (element transpose)
        {
            const __bf16* vsrc = Vc + (kvbase + kv0 + sr) * CD + ssg;
            __bf16 tmp[16];
            ((uint4*)tmp)[0] = ((const uint4*)vsrc)[0];
            ((uint4*)tmp)[1] = ((const uint4*)vsrc)[1];
            #pragma unroll
            for (int e = 0; e < 16; ++e) lVt[(ssg + e) * 32 + sr] = tmp[e];
        }
        wait_async0();
        __syncthreads();

        // causal: this wave's rows all precede the tile? skip compute
        // (wave-uniform branch => EXEC stays all-ones for WMMA)
        if (kv0 < qrow0 + 16) {
            // S = Q * K^T  (16 x 32)
            v8f sc[2] = {};
            #pragma unroll
            for (int j = 0; j < 2; ++j)
                #pragma unroll
                for (int kk = 0; kk < 4; ++kk) {
                    v16bf bk = load_frag(lK + (j * 16) * 128 + kk * 32, 128);
                    sc[j] = __builtin_amdgcn_wmma_f32_16x16x32_bf16(
                        false, aq[kk], false, bk, (short)0, sc[j], false, false);
                }

            // scale + causal mask + online softmax, write P to LDS
            #pragma unroll
            for (int v = 0; v < 8; ++v) {
                int m = qrow0 + 8 * hi + v;
                float x0 = sc[0][v] * scale, x1 = sc[1][v] * scale;
                if (kv0 + col      > m) x0 = -1e30f;
                if (kv0 + 16 + col > m) x1 = -1e30f;
                float rm = fmaxf(x0, x1);
                #pragma unroll
                for (int msk = 8; msk; msk >>= 1) rm = fmaxf(rm, __shfl_xor(rm, msk, 32));
                float mnew = fmaxf(mrow[v], rm);
                float fac  = __expf(mrow[v] - mnew);
                float p0 = __expf(x0 - mnew), p1 = __expf(x1 - mnew);
                float rs = p0 + p1;
                #pragma unroll
                for (int msk = 8; msk; msk >>= 1) rs += __shfl_xor(rs, msk, 32);
                lrow[v] = lrow[v] * fac + rs;
                mrow[v] = mnew;
                #pragma unroll
                for (int c = 0; c < 8; ++c) o[c][v] *= fac;
                lP[wave][(v + 8 * hi) * 32 + col]      = (__bf16)p0;
                lP[wave][(v + 8 * hi) * 32 + 16 + col] = (__bf16)p1;
            }

            // O += P * V
            v16bf ap = load_frag(lP[wave], 32);
            #pragma unroll
            for (int c = 0; c < 8; ++c) {
                v16bf bv = load_frag(lVt + (c * 16) * 32, 32);
                o[c] = __builtin_amdgcn_wmma_f32_16x16x32_bf16(
                    false, ap, false, bv, (short)0, o[c], false, false);
            }
        }
        __syncthreads();
    }

    // epilogue: 1/l, sigmoid(gate), store [B,S,H*D] bf16
    #pragma unroll
    for (int v = 0; v < 8; ++v) {
        int   sg   = qrow0 + 8 * hi + v;
        float invl = 1.0f / lrow[v];
        size_t grow = ((size_t)(b * CH + h) * CS + sg) * CD;
        size_t orow = ((size_t)(b * CS + sg) * CH + h) * CD;
        #pragma unroll
        for (int c = 0; c < 8; ++c) {
            int d = c * 16 + col;
            float g   = Gate[grow + d];
            float sig = 1.0f / (1.0f + __expf(-g));
            Out[orow + d] = (__bf16)(o[c][v] * invl * sig);
        }
    }
}

// ---------------------------------------------------------------- launch
extern "C" void kernel_launch(void* const* d_in, const int* in_sizes, int n_in,
                              void* d_out, int out_size, void* d_ws, size_t ws_size,
                              hipStream_t stream) {
    const float* x    = (const float*)d_in[0];
    const float* Wq   = (const float*)d_in[1];
    const float* Wk   = (const float*)d_in[2];
    const float* Wv   = (const float*)d_in[3];
    const float* Wo   = (const float*)d_in[4];
    const float* qn_w = (const float*)d_in[5];
    const float* kn_w = (const float*)d_in[6];

    char* ws = (char*)d_ws;
    size_t off = 0;
    auto carve = [&](size_t bytes) { size_t o = off; off += (bytes + 255) & ~(size_t)255; return o; };

    __bf16* xb    = (__bf16*)(ws + carve((size_t)CM * CHID * 2));
    __bf16* wqT   = (__bf16*)(ws + carve((size_t)CNQ * CHID * 2));
    __bf16* wkT   = (__bf16*)(ws + carve((size_t)CNK * CHID * 2));
    __bf16* wvT   = (__bf16*)(ws + carve((size_t)CNK * CHID * 2));
    __bf16* woT   = (__bf16*)(ws + carve((size_t)CHID * CHID * 2));
    float*  qraw  = (float*) (ws + carve((size_t)CM * CNQ * 4));
    float*  kraw  = (float*) (ws + carve((size_t)CM * CNK * 4));
    float*  vraw  = (float*) (ws + carve((size_t)CM * CNK * 4));
    __bf16* qb    = (__bf16*)(ws + carve((size_t)CB * CH * CS * CD * 2));
    float*  gate  = (float*) (ws + carve((size_t)CB * CH * CS * CD * 4));
    __bf16* kb    = (__bf16*)(ws + carve((size_t)CB * CKV * CS * CD * 2));
    __bf16* vb    = (__bf16*)(ws + carve((size_t)CB * CKV * CS * CD * 2));
    __bf16* attnb = (__bf16*)(ws + carve((size_t)CM * CH * CD * 2));

    // 1) precision conversion / weight transposes
    cast_bf16_kernel<<<(CM * CHID) / 256, 256, 0, stream>>>(x, xb, CM * CHID);
    cast_transpose_kernel<<<(CHID * CNQ) / 256, 256, 0, stream>>>(Wq, wqT, CHID, CNQ);
    cast_transpose_kernel<<<(CHID * CNK) / 256, 256, 0, stream>>>(Wk, wkT, CHID, CNK);
    cast_transpose_kernel<<<(CHID * CNK) / 256, 256, 0, stream>>>(Wv, wvT, CHID, CNK);
    cast_transpose_kernel<<<(CHID * CHID) / 256, 256, 0, stream>>>(Wo, woT, CHID, CHID);

    // 2) QKV projection GEMMs (bf16 WMMA, fp32 accumulate)
    gemm_bf16_wmma<<<dim3(CNQ / 128, CM / 128), 256, 0, stream>>>(xb, wqT, qraw, CM, CNQ, CHID);
    gemm_bf16_wmma<<<dim3(CNK / 128, CM / 128), 256, 0, stream>>>(xb, wkT, kraw, CM, CNK, CHID);
    gemm_bf16_wmma<<<dim3(CNK / 128, CM / 128), 256, 0, stream>>>(xb, wvT, vraw, CM, CNK, CHID);

    // 3) RMSNorm + RoPE + head-major relayout
    qkv_postproc_kernel<<<CM, 256, 0, stream>>>(qraw, kraw, vraw, qn_w, kn_w,
                                                qb, gate, kb, vb);

    // 4) causal flash attention + sigmoid gate
    flash_attn_wmma<<<dim3(CS / 128, CH, CB), 256, 0, stream>>>(qb, kb, vb, gate, attnb);

    // 5) output projection
    gemm_bf16_wmma<<<dim3(CHID / 128, CM / 128), 256, 0, stream>>>(attnb, woT, (float*)d_out,
                                                                   CM, CHID, CHID);
}
